// ResponseMapNet_23106924053147
// MI455X (gfx1250) — compile-verified, matrix operand order checked by reading, not run on hardware
//
#include <hip/hip_runtime.h>
#include <hip/hip_bf16.h>

typedef __attribute__((ext_vector_type(16))) _Float16 v16h;
typedef __attribute__((ext_vector_type(8)))  _Float16 h8;
typedef __attribute__((ext_vector_type(4)))  _Float16 h4;
typedef __attribute__((ext_vector_type(8)))  float    v8f;

#define BN_EPS 1e-5f
#define NEG_SLOPE 0.01f

// ---- workspace layout (bytes) ----
#define WS_W1FRAG   0          // 4kt*2nt*32lane*16 f16  = 8192 B   (K padded 108->128)
#define WS_W2FRAG   8192       // 27kt*4nt*32*16 f16     = 110592 B (K=864 exactly)
#define WS_FC1FRAG  118784     // 16kt*8nt*32*16 f16     = 131072 B (K=512, B = fcw1^T)
#define WS_S1       249856     // 32 f32  g1/sqrt(v1+eps)
#define WS_T1       249984     // 32 f32  b1 - m1*s1
#define WS_S2       250112     // 64 f32
#define WS_T2       250368     // 64 f32
#define WS_H2SWZ    262144     // 4096*512 f16 = 4 MB, A-fragment-swizzled FC input

static __device__ __forceinline__ v8f wmma_f16(v16h a, v16h b, v8f c) {
    return __builtin_amdgcn_wmma_f32_16x16x32_f16(false, a, false, b, (short)0, c, false, false);
}

// tap -> padded-volume linear delta (compile-time folded after unrolling)
__device__ __constant__ const int DELTA1[27] = {   // dz*100 + dy*10 + dx  (10^3 pad)
      0,   1,   2,  10,  11,  12,  20,  21,  22,
    100, 101, 102, 110, 111, 112, 120, 121, 122,
    200, 201, 202, 210, 211, 212, 220, 221, 222 };
__device__ __constant__ const int DELTA2[27] = {   // dz*36 + dy*6 + dx    (6^3 pad)
      0,  1,  2,  6,  7,  8, 12, 13, 14,
     36, 37, 38, 42, 43, 44, 48, 49, 50,
     72, 73, 74, 78, 79, 80, 84, 85, 86 };

// ============================================================================
// Kernel 0: pre-pack weights into WMMA B-fragment order (f16) + fold BN params
// B-fragment element map (16-bit, 16x16x32): lane L -> column n = nt*16 + (L%16),
// k-local = (v/4)*16 + (L/16)*8 + (v%4)*2 + e  for vector slot idx16 = v*2+e.
// ============================================================================
__global__ __launch_bounds__(256) void prepack_kernel(
    const float* __restrict__ W1, const float* __restrict__ W2,
    const float* __restrict__ fcw1,
    const float* __restrict__ g1, const float* __restrict__ b1,
    const float* __restrict__ m1, const float* __restrict__ v1,
    const float* __restrict__ g2, const float* __restrict__ b2,
    const float* __restrict__ m2, const float* __restrict__ v2,
    unsigned char* __restrict__ ws) {
    const int gid = blockIdx.x * 256 + threadIdx.x;
    const int N1 = 4 * 2 * 512;     // w1frag elems  (4096)
    const int N2 = 27 * 4 * 512;    // w2frag elems  (55296)
    const int N3 = 16 * 8 * 512;    // fc1frag elems (65536)
    _Float16* w1f = (_Float16*)(ws + WS_W1FRAG);
    _Float16* w2f = (_Float16*)(ws + WS_W2FRAG);
    _Float16* fcf = (_Float16*)(ws + WS_FC1FRAG);

    if (gid < N1 + N2 + N3) {
        int set, local, NT;
        if (gid < N1)            { set = 0; local = gid;            NT = 2; }
        else if (gid < N1 + N2)  { set = 1; local = gid - N1;       NT = 4; }
        else                     { set = 2; local = gid - N1 - N2;  NT = 8; }
        const int idx16 = local & 15;
        const int lane  = (local >> 4) & 31;
        const int fi    = local >> 9;
        const int kt = fi / NT, nt = fi % NT;
        const int v = idx16 >> 1, e = idx16 & 1;
        const int half = lane >> 4, col = lane & 15;
        const int kl = (v >> 2) * 16 + half * 8 + (v & 3) * 2 + e;
        const int k = kt * 32 + kl;
        const int n = nt * 16 + col;
        if (set == 0) {
            // W1: (3,3,3,4,32) flat = [K=108][N=32]; pad K to 128 with zeros
            float val = (k < 108) ? W1[k * 32 + n] : 0.0f;
            w1f[local] = (_Float16)val;
        } else if (set == 1) {
            // W2: (3,3,3,32,64) flat = [K=864][N=64]
            w2f[local] = (_Float16)W2[k * 64 + n];
        } else {
            // fcw1: [128,512]; B[k][n] = fcw1[n][k]
            fcf[local] = (_Float16)fcw1[n * 512 + k];
        }
    } else {
        int t = gid - (N1 + N2 + N3);
        float* s1 = (float*)(ws + WS_S1); float* t1 = (float*)(ws + WS_T1);
        float* s2 = (float*)(ws + WS_S2); float* t2 = (float*)(ws + WS_T2);
        if (t < 32) {
            float s = g1[t] * rsqrtf(v1[t] + BN_EPS);
            s1[t] = s; t1[t] = b1[t] - m1[t] * s;
        } else if (t < 96) {
            int c = t - 32;
            float s = g2[c] * rsqrtf(v2[c] + BN_EPS);
            s2[c] = s; t2[c] = b2[c] - m2[c] * s;
        }
    }
}

// ============================================================================
// Kernel 1: fully fused per-batch pipeline, one workgroup (8 wave32) per batch.
//   conv1 (implicit GEMM M=512,N=32,K=128) + mask/BN/ReLU -> LDS
//   maxpool 2^3 -> zero-haloed LDS [6^3][32]
//   conv2 (implicit GEMM M=64,N=64,K=27x32) + mask/BN/ReLU -> LDS
//   maxpool 2^3 -> fragment-swizzled f16 [B,512] in workspace
// LDS union:  xpad [0,8000) | act [8016,40784) | h1pad [40784,54608)
//             act2 reuses [0,8192) after xpad is dead.
// xpad[0..3] (halo corner voxel) is guaranteed zero -> dead taps load offset 0.
// ============================================================================
__global__ __launch_bounds__(256) void conv_fused_kernel(
    const float* __restrict__ x, const unsigned char* __restrict__ mask,
    unsigned char* __restrict__ ws) {
    __shared__ __align__(16) unsigned char smem[54608];
    __shared__ unsigned char mask1[512];
    __shared__ unsigned char mask2[64];
    _Float16* xpad  = (_Float16*)(smem);            // [10][10][10][4]
    _Float16* act   = (_Float16*)(smem + 8016);     // [512 vox][32 ch]
    _Float16* h1pad = (_Float16*)(smem + 40784);    // [6][6][6][32]
    _Float16* act2  = (_Float16*)(smem);            // [64 vox][64 ch] (reuse)

    const int tid  = threadIdx.x;
    const int b    = blockIdx.x;
    const int lane = tid & 31, wv = tid >> 5;
    const int half = lane >> 4, col = lane & 15;

    const v16h*  w1f   = (const v16h*)(ws + WS_W1FRAG);
    const v16h*  w2f   = (const v16h*)(ws + WS_W2FRAG);
    const float* s1    = (const float*)(ws + WS_S1);
    const float* t1    = (const float*)(ws + WS_T1);
    const float* s2    = (const float*)(ws + WS_S2);
    const float* t2    = (const float*)(ws + WS_T2);
    _Float16*    h2swz = (_Float16*)(ws + WS_H2SWZ);

    // ---- phase 0: zero halo buffers, stage mask ----
    {
        unsigned int* z0 = (unsigned int*)smem;               // xpad: 8000 B
        for (int i = tid; i < 2000; i += 256) z0[i] = 0u;
        unsigned int* z1 = (unsigned int*)(smem + 40784);     // h1pad: 13824 B
        for (int i = tid; i < 3456; i += 256) z1[i] = 0u;
        const unsigned int* mg = (const unsigned int*)(mask + (size_t)b * 512);
        unsigned int* ml = (unsigned int*)mask1;
        for (int i = tid; i < 128; i += 256) ml[i] = mg[i];
    }
    __syncthreads();

    // ---- phase 1: x[b] (fp32, [4][8][8][8]) -> f16 haloed [10^3][4] ----
    {
        const float* xb = x + (size_t)b * 2048;
        for (int i = tid; i < 2048; i += 256) {
            int c = i >> 9, vox = i & 511;
            int d = vox >> 6, h = (vox >> 3) & 7, w = vox & 7;
            xpad[(((d + 1) * 10 + (h + 1)) * 10 + (w + 1)) * 4 + c] = (_Float16)xb[i];
        }
    }
    __syncthreads();

    // ---- phase 2: conv1 GEMM. A[m][k]: m=voxel, k=tap*4+cin (K padded to 128).
    // A-fragment for lane = 4 taps x 4 contiguous cin -> four 8-byte LDS loads,
    // all offsets derived from compile-time deltas; dead taps -> zero corner.
    {
        float sa = s1[col], ta = t1[col], sb = s1[16 + col], tb = t1[16 + col];
        for (int mi = 0; mi < 4; ++mi) {
            const int mt = mi * 8 + wv;
            const int m = mt * 16 + col;
            const int d = m >> 6, h = (m >> 3) & 7, w = m & 7;
            const int base0 = d * 100 + h * 10 + w;   // padded-volume voxel base
            v8f acc0 = {}, acc1 = {};
            #pragma unroll
            for (int kt = 0; kt < 4; ++kt) {
                union { v16h v; h4 q[4]; } A;
                #pragma unroll
                for (int j = 0; j < 4; ++j) {
                    const int J  = (j < 2) ? j : (j + 2);   // 0,1,4,5
                    const int ta0 = kt * 8 + J;             // half==0 tap
                    int off;
                    if (ta0 + 2 < 27) {        // both halves valid
                        off = (base0 + (half ? DELTA1[ta0 + 2] : DELTA1[ta0])) * 4;
                    } else if (ta0 < 27) {     // only half==0 valid (tap 25)
                        off = half ? 0 : (base0 + DELTA1[ta0]) * 4;
                    } else {                   // padding region -> zero corner
                        off = 0;
                    }
                    A.q[j] = *(const h4*)(xpad + off);
                }
                v16h bf0 = w1f[kt * 64 + lane];
                v16h bf1 = w1f[kt * 64 + 32 + lane];
                acc0 = wmma_f16(A.v, bf0, acc0);
                acc1 = wmma_f16(A.v, bf1, acc1);
            }
            // epilogue: D layout lane->n, vgpr->m; mask * relu(acc*s + t)
            const int mbase = mt * 16 + half * 8;
            #pragma unroll
            for (int v = 0; v < 8; ++v) {
                int mm = mbase + v;
                float mk = mask1[mm] ? 1.0f : 0.0f;
                float y0 = fmaxf(acc0[v] * sa + ta, 0.0f) * mk;
                float y1 = fmaxf(acc1[v] * sb + tb, 0.0f) * mk;
                act[mm * 32 + col]      = (_Float16)y0;
                act[mm * 32 + 16 + col] = (_Float16)y1;
            }
        }
    }
    __syncthreads();

    // ---- phase 3: maxpool 8^3 -> 4^3 into haloed h1pad; pooled mask ----
    for (int i = tid; i < 2048; i += 256) {
        int ch = i & 31, ov = i >> 5;
        int d2 = ov >> 4, h2 = (ov >> 2) & 3, w2 = ov & 3;
        float mx = 0.0f;
        #pragma unroll
        for (int a = 0; a < 2; ++a)
            #pragma unroll
            for (int bb = 0; bb < 2; ++bb)
                #pragma unroll
                for (int cc = 0; cc < 2; ++cc)
                    mx = fmaxf(mx, (float)act[(((d2*2+a)*8 + h2*2+bb)*8 + w2*2+cc)*32 + ch]);
        h1pad[(((d2 + 1) * 6 + (h2 + 1)) * 6 + (w2 + 1)) * 32 + ch] = (_Float16)mx;
    }
    if (tid < 64) {
        int d2 = tid >> 4, h2 = (tid >> 2) & 3, w2 = tid & 3;
        unsigned char anym = 0;
        #pragma unroll
        for (int a = 0; a < 2; ++a)
            #pragma unroll
            for (int bb = 0; bb < 2; ++bb)
                #pragma unroll
                for (int cc = 0; cc < 2; ++cc)
                    anym |= mask1[((d2*2+a)*8 + h2*2+bb)*8 + w2*2+cc];
        mask2[tid] = anym;
    }
    __syncthreads();

    // ---- phase 4: conv2 GEMM. K tiles == 27 taps (kl == cin, contiguous).
    // Fully unrolled taps: all DS/VMEM offsets are immediates.
    {
        const int mt = wv >> 1, nh = wv & 1;
        const int n0 = nh * 32 + col, n1 = n0 + 16;
        float sa = s2[n0], ta = t2[n0], sb = s2[n1], tb = t2[n1];
        const int m = mt * 16 + col;
        const int d = m >> 4, h = (m >> 2) & 3, w = m & 3;
        const int abase = (d * 36 + h * 6 + w) * 32 + half * 8;  // f16 units
        const v16h* w2base = w2f + nh * 64 + lane;               // + kt*128 (+32)
        v8f acc0 = {}, acc1 = {};
        #pragma unroll
        for (int kt = 0; kt < 27; ++kt) {
            const int doff = DELTA2[kt] * 32;
            union { v16h v; h8 o[2]; } A;
            A.o[0] = *(const h8*)(h1pad + abase + doff);
            A.o[1] = *(const h8*)(h1pad + abase + doff + 16);
            v16h bf0 = w2base[kt * 128];
            v16h bf1 = w2base[kt * 128 + 32];
            acc0 = wmma_f16(A.v, bf0, acc0);
            acc1 = wmma_f16(A.v, bf1, acc1);
        }
        const int mbase = mt * 16 + half * 8;
        #pragma unroll
        for (int v = 0; v < 8; ++v) {
            int mm = mbase + v;
            float mk = mask2[mm] ? 1.0f : 0.0f;
            float y0 = fmaxf(acc0[v] * sa + ta, 0.0f) * mk;
            float y1 = fmaxf(acc1[v] * sb + tb, 0.0f) * mk;
            act2[mm * 64 + n0] = (_Float16)y0;
            act2[mm * 64 + n1] = (_Float16)y1;
        }
    }
    __syncthreads();

    // ---- phase 5: maxpool 4^3 -> 2^3; k = c*8 + voxel (matches flatten);
    // scatter into FC A-fragment-swizzled layout ----
    for (int i = tid; i < 512; i += 256) {
        int c = i >> 3, pv = i & 7;
        int d = pv >> 2, h = (pv >> 1) & 1, w = pv & 1;
        float mx = 0.0f;
        #pragma unroll
        for (int a = 0; a < 2; ++a)
            #pragma unroll
            for (int bb = 0; bb < 2; ++bb)
                #pragma unroll
                for (int cc = 0; cc < 2; ++cc)
                    mx = fmaxf(mx, (float)act2[(((d*2+a)*4 + h*2+bb)*4 + w*2+cc)*64 + c]);
        int k = c * 8 + pv;
        int kt = k >> 5, kl = k & 31;
        int g = kl >> 4, low = kl & 15, hf = low >> 3, r3 = low & 7;
        int v = g * 4 + (r3 >> 1), e = r3 & 1;
        int ln = hf * 16 + (b & 15);
        size_t pos = ((size_t)((b >> 4) * 16 + kt) * 32 + ln) * 16 + (v * 2 + e);
        h2swz[pos] = (_Float16)mx;
    }
}

// ============================================================================
// Kernel 2: FC1 (GEMM M=batch,N=128,K=512) + bias + LeakyReLU + FC2 (dot 128)
// One wave per 16-batch row-block; 128 batches per workgroup.
// ============================================================================
__global__ __launch_bounds__(256) void fc_kernel(
    const unsigned char* __restrict__ ws,
    const float* __restrict__ fcb1, const float* __restrict__ fcw2,
    const float* __restrict__ fcb2, float* __restrict__ out) {
    __shared__ _Float16 h3[128 * 128];  // 32 KB
    const int tid  = threadIdx.x;
    const int lane = tid & 31, wv = tid >> 5;
    const int half = lane >> 4, col = lane & 15;
    const v16h* a_frags = (const v16h*)(ws + WS_H2SWZ);
    const v16h* b_frags = (const v16h*)(ws + WS_FC1FRAG);
    const int rb = blockIdx.x * 8 + wv;   // row-block of 16 batches

    v8f acc[8];
    #pragma unroll
    for (int nt = 0; nt < 8; ++nt) { v8f z = {}; acc[nt] = z; }

    for (int kt = 0; kt < 16; ++kt) {
        v16h a = a_frags[(rb * 16 + kt) * 32 + lane];
        #pragma unroll
        for (int nt = 0; nt < 8; ++nt) {
            v16h bf = b_frags[(kt * 8 + nt) * 32 + lane];
            acc[nt] = wmma_f16(a, bf, acc[nt]);
        }
    }
    #pragma unroll
    for (int nt = 0; nt < 8; ++nt) {
        int n = nt * 16 + col;
        float fb = fcb1[n];
        #pragma unroll
        for (int v = 0; v < 8; ++v) {
            float val = acc[nt][v] + fb;
            val = (val >= 0.0f) ? val : NEG_SLOPE * val;
            h3[(wv * 16 + half * 8 + v) * 128 + n] = (_Float16)val;
        }
    }
    __syncthreads();
    if (tid < 128) {
        float s = fcb2[0];
        #pragma unroll 8
        for (int j = 0; j < 128; ++j)
            s += (float)h3[tid * 128 + j] * fcw2[j];
        out[blockIdx.x * 128 + tid] = s;
    }
}

extern "C" void kernel_launch(void* const* d_in, const int* in_sizes, int n_in,
                              void* d_out, int out_size, void* d_ws, size_t ws_size,
                              hipStream_t stream) {
    (void)in_sizes; (void)n_in; (void)out_size; (void)ws_size;
    const float* x           = (const float*)d_in[0];
    const unsigned char* msk = (const unsigned char*)d_in[1];
    const float* W1   = (const float*)d_in[2];
    const float* g1   = (const float*)d_in[3];
    const float* b1   = (const float*)d_in[4];
    const float* m1   = (const float*)d_in[5];
    const float* v1   = (const float*)d_in[6];
    const float* W2   = (const float*)d_in[7];
    const float* g2   = (const float*)d_in[8];
    const float* b2   = (const float*)d_in[9];
    const float* m2   = (const float*)d_in[10];
    const float* v2   = (const float*)d_in[11];
    const float* fcw1 = (const float*)d_in[12];
    const float* fcb1 = (const float*)d_in[13];
    const float* fcw2 = (const float*)d_in[14];
    const float* fcb2 = (const float*)d_in[15];
    unsigned char* ws = (unsigned char*)d_ws;
    float* out = (float*)d_out;

    // 4096 + 55296 + 65536 frag elems + 96 BN lanes = 125024 -> 489 blocks
    prepack_kernel<<<489, 256, 0, stream>>>(W1, W2, fcw1, g1, b1, m1, v1,
                                            g2, b2, m2, v2, ws);
    conv_fused_kernel<<<4096, 256, 0, stream>>>(x, msk, ws);
    fc_kernel<<<32, 256, 0, stream>>>(ws, fcb1, fcw2, fcb2, out);
}
